// ModelMamba_78658031059336
// MI455X (gfx1250) — compile-verified, matrix operand order checked by reading, not run on hardware
//
#include <hip/hip_runtime.h>

#define B_      16
#define L_      2048
#define D_TOK   256
#define D_TIS   64
#define D_MODEL 320
#define D_INNER 640
#define D_STATE 16
#define D_CONV  4
#define DT_RANK 20
#define H_OUT   128
#define VOCAB   65
#define XD      (DT_RANK + 2 * D_STATE)   // 52

typedef __attribute__((ext_vector_type(16))) __bf16 v16bf;
typedef __attribute__((ext_vector_type(8)))  float  v8f;

static __device__ __forceinline__ unsigned short f32_to_bf16(float f) {
  unsigned int u = __float_as_uint(f);
  u += 0x7fffu + ((u >> 16) & 1u);           // round-to-nearest-even
  return (unsigned short)(u >> 16);
}
static __device__ __forceinline__ float sigmoidf_(float x) { return 1.f / (1.f + __expf(-x)); }
static __device__ __forceinline__ float softplusf_(float x) { return (x > 20.f) ? x : log1pf(__expf(x)); }

// Async global->LDS copy of one contiguous region in 16B chunks (CDNA5 path,
// tracked by ASYNCcnt). ldsOff = LDS byte offset, g = global base.
static __device__ __forceinline__ void async_copy_b128(unsigned ldsOff, const char* g) {
  asm volatile("global_load_async_to_lds_b128 %0, %1, off" :: "v"(ldsOff), "v"(g) : "memory");
}
static __device__ __forceinline__ void async_wait0() {
  asm volatile("s_wait_asynccnt 0x0" ::: "memory");
}

// ---------------------------------------------------------------- embeddings
__global__ void __launch_bounds__(128) norms_kernel(
    const float* __restrict__ seq_W, const float* __restrict__ tis_W,
    float* __restrict__ seq_scale, float* __restrict__ tis_scale) {
  const int t = threadIdx.x;
  if (t < VOCAB) {
    float ss = 0.f;
    for (int i = 0; i < D_TOK; ++i) { float v = seq_W[t * D_TOK + i]; ss += v * v; }
    seq_scale[t] = fminf(1.f, 2.f / fmaxf(sqrtf(ss), 1e-12f));
  } else if (t < VOCAB + 30) {
    const int r = t - VOCAB;
    float ss = 0.f;
    for (int i = 0; i < D_TIS; ++i) { float v = tis_W[r * D_TIS + i]; ss += v * v; }
    tis_scale[r] = fminf(1.f, 2.f / fmaxf(sqrtf(ss), 1e-12f));
  }
}

__global__ void __launch_bounds__(320) wcast_kernel(const float* __restrict__ W,
                                                    unsigned short* __restrict__ Wbf) {
  const int i = blockIdx.x * 320 + threadIdx.x;   // covers 640*320 elements (rows 0..639 = Wx)
  Wbf[i] = f32_to_bf16(W[i]);
}

__global__ void __launch_bounds__(320) embed_kernel(
    const int* __restrict__ rna, const int* __restrict__ tissue_id,
    const float* __restrict__ seq_W, const float* __restrict__ tis_W,
    const float* __restrict__ seq_scale, const float* __restrict__ tis_scale,
    unsigned short* __restrict__ Xbf) {
  const int p = blockIdx.x;          // b*L + t
  const int b = p / L_;
  const int c = threadIdx.x;
  const int idx = rna[p];
  float v;
  if (c < D_TOK) {
    v = seq_W[idx * D_TOK + c] * seq_scale[idx];
  } else {
    const int ti = tissue_id[b];
    v = tis_W[ti * D_TIS + (c - D_TOK)] * tis_scale[ti];
  }
  if (idx == 0) v = 0.f;             // padding mask zeroes the whole feature
  Xbf[(size_t)p * D_MODEL + c] = f32_to_bf16(v);
}

// --------------------------------------------- in_proj (xm half) via WMMA bf16
// xm_raw[M,N] = X[M,K] * Wx[N,K]^T ; M=32768, K=320, N=640
__global__ void __launch_bounds__(256) inproj_wmma_kernel(
    const unsigned short* __restrict__ Xbf,   // [M][320] bf16
    const unsigned short* __restrict__ Wbf,   // [640][320] bf16
    float* __restrict__ xm_raw) {             // [M][640]
  __shared__ unsigned int ldsA[16 * 160];     // 16 rows x 320 bf16 = 10240 B, contiguous in X
  const int tid = threadIdx.x;
  const int mbase = blockIdx.y * 16;
  {
    // Async-stage the whole contiguous A tile: 640 x 16B chunks over 256 lanes.
    const unsigned lbase = (unsigned)(size_t)&ldsA[0];
    const char* gbase = (const char*)(Xbf + (size_t)mbase * D_MODEL);
    for (int i = tid; i < 640; i += 256)
      async_copy_b128(lbase + i * 16, gbase + (size_t)i * 16);
    async_wait0();
  }
  __syncthreads();

  const int w    = tid >> 5;        // 8 waves -> 8 N-tiles per block
  const int lane = tid & 31;
  const int half = lane >> 4;
  const int mrow = lane & 15;
  const int ncol = lane & 15;
  const int n    = blockIdx.x * 128 + w * 16 + ncol;  // weight row (output channel)
  const unsigned int* Wg = (const unsigned int*)Wbf;

  v8f c = {};
  #pragma unroll
  for (int kk = 0; kk < 10; ++kk) {           // K = 320 = 10 * 32
    union { unsigned int u[8]; v16bf v; } a, bfr;
    const int k0 = kk * 32;
    #pragma unroll
    for (int j = 0; j < 8; ++j) {
      // ISA 7.12.2 16-bit A layout: lanes 0-15 hold K {0..7,16..23}, lanes 16-31 {8..15,24..31}
      const int koff = 2 * j + ((j >= 4) ? 8 : 0) + 8 * half;
      a.u[j]   = ldsA[mrow * 160 + ((k0 + koff) >> 1)];
      bfr.u[j] = Wg[(size_t)n * 160 + ((k0 + koff) >> 1)];
    }
    c = __builtin_amdgcn_wmma_f32_16x16x32_bf16(false, a.v, false, bfr.v,
                                                (short)0, c, false, false);
  }
  #pragma unroll
  for (int j = 0; j < 8; ++j) {               // C/D layout: M = j + 8*half, N = ncol
    const int row = mbase + j + 8 * half;
    xm_raw[(size_t)row * D_INNER + blockIdx.x * 128 + w * 16 + ncol] = c[j];
  }
}

// ---------------------------------------------------- depthwise conv4 + silu
__global__ void __launch_bounds__(640) conv_silu_kernel(
    const float* __restrict__ xm_raw, const float* __restrict__ conv_w,
    const float* __restrict__ conv_b, float* __restrict__ xm) {
  const int p = blockIdx.x;      // b*L + t
  const int t = p % L_;
  const int d = threadIdx.x;
  float acc = conv_b[d];
  #pragma unroll
  for (int j = 0; j < D_CONV; ++j) {
    const int tt = t - (D_CONV - 1) + j;      // causal
    if (tt >= 0) acc += conv_w[d * D_CONV + j] * xm_raw[(size_t)(p - t + tt) * D_INNER + d];
  }
  xm[(size_t)p * D_INNER + d] = acc * sigmoidf_(acc);
}

// ------------------------------------------- x_proj: dt(20) | B(16) | C(16)
__global__ void __launch_bounds__(64) xproj_kernel(
    const float* __restrict__ xm, const float* __restrict__ x_proj_W,
    float* __restrict__ dtBC) {
  __shared__ float row[D_INNER];              // 2560 B contiguous
  const int p = blockIdx.x;
  {
    const unsigned lbase = (unsigned)(size_t)&row[0];
    const char* gbase = (const char*)(xm + (size_t)p * D_INNER);
    for (int i = threadIdx.x; i < D_INNER / 4; i += 64)   // 160 x 16B chunks
      async_copy_b128(lbase + i * 16, gbase + (size_t)i * 16);
    async_wait0();
  }
  __syncthreads();
  const int o = threadIdx.x;
  if (o < XD) {
    const float* wrow = x_proj_W + (size_t)o * D_INNER;
    float acc = 0.f;
    for (int i = 0; i < D_INNER; ++i) acc += row[i] * wrow[i];
    dtBC[(size_t)p * XD + o] = acc;
  }
}

// ------------------------------------------------------------ selective scan
// One lane owns one (b,d); h[16] lives in registers; only y at t_last is kept.
__global__ void __launch_bounds__(128) scan_kernel(
    const float* __restrict__ dtBC, const float* __restrict__ xm,
    const float* __restrict__ dt_proj_W, const float* __restrict__ dt_proj_b,
    const float* __restrict__ A_log, const float* __restrict__ D_skip,
    const int* __restrict__ seq_lengths, float* __restrict__ y_last) {
  const int b = blockIdx.x;
  const int d = blockIdx.y * 128 + threadIdx.x;
  float h[D_STATE], Av[D_STATE], w[DT_RANK];
  #pragma unroll
  for (int s = 0; s < D_STATE; ++s) { h[s] = 0.f; Av[s] = -__expf(A_log[d * D_STATE + s]); }
  #pragma unroll
  for (int r = 0; r < DT_RANK; ++r) w[r] = dt_proj_W[d * DT_RANK + r];
  const float bias  = dt_proj_b[d];
  const float dsk   = D_skip[d];
  const int   tl    = seq_lengths[b] - 1;
  const float* rowp = dtBC + (size_t)b * L_ * XD;
  const float* up   = xm + (size_t)b * L_ * D_INNER + d;

  for (int t = 0; t <= tl; ++t) {
    const float* row = rowp + (size_t)t * XD;
    __builtin_prefetch(row + XD, 0, 1);       // global_prefetch_b8
    float x = bias;
    #pragma unroll
    for (int r = 0; r < DT_RANK; ++r) x += row[r] * w[r];
    const float delta = softplusf_(x);
    const float u  = up[(size_t)t * D_INNER];
    const float du = delta * u;
    #pragma unroll
    for (int s = 0; s < D_STATE; ++s)
      h[s] = __expf(delta * Av[s]) * h[s] + du * row[DT_RANK + s];
    if (t == tl) {
      float y = u * dsk;
      #pragma unroll
      for (int s = 0; s < D_STATE; ++s) y += h[s] * row[DT_RANK + D_STATE + s];
      y_last[b * D_INNER + d] = y;
    }
  }
}

// ----------------------------------------------- final head (last step only)
__global__ void __launch_bounds__(640) head_kernel(
    const int* __restrict__ rna, const int* __restrict__ tissue_id,
    const int* __restrict__ seq_lengths,
    const float* __restrict__ seq_W, const float* __restrict__ tis_W,
    const float* __restrict__ seq_scale, const float* __restrict__ tis_scale,
    const float* __restrict__ in_proj_W, const float* __restrict__ y_last,
    const float* __restrict__ out_proj_W,
    const float* __restrict__ p1_W, const float* __restrict__ p1_b,
    const float* __restrict__ p2_W, const float* __restrict__ p2_b,
    float* __restrict__ out) {
  __shared__ float xl[D_MODEL];
  __shared__ float yb[D_INNER];
  __shared__ float o3[D_MODEL];
  __shared__ float h1[H_OUT];
  const int b   = blockIdx.x;
  const int tl  = seq_lengths[b] - 1;
  const int idx = rna[b * L_ + tl];
  const int tid = threadIdx.x;
  if (tid < D_MODEL) {                        // recompute x at t_last in f32
    float v;
    if (tid < D_TOK) v = seq_W[idx * D_TOK + tid] * seq_scale[idx];
    else { const int ti = tissue_id[b]; v = tis_W[ti * D_TIS + (tid - D_TOK)] * tis_scale[ti]; }
    xl[tid] = (idx == 0) ? 0.f : v;
  }
  __syncthreads();
  {                                           // z gate: rows 640..1279 of in_proj_W
    const float* wrow = in_proj_W + (size_t)(D_INNER + tid) * D_MODEL;
    float z = 0.f;
    for (int i = 0; i < D_MODEL; ++i) z += wrow[i] * xl[i];
    yb[tid] = y_last[b * D_INNER + tid] * (z * sigmoidf_(z));
  }
  __syncthreads();
  if (tid < D_MODEL) {
    const float* wrow = out_proj_W + (size_t)tid * D_INNER;
    float acc = 0.f;
    for (int i = 0; i < D_INNER; ++i) acc += wrow[i] * yb[i];
    o3[tid] = acc;
  }
  __syncthreads();
  if (tid < H_OUT) {
    const float* wrow = p1_W + (size_t)tid * D_MODEL;
    float acc = p1_b[tid];
    for (int i = 0; i < D_MODEL; ++i) acc += wrow[i] * o3[i];
    h1[tid] = fmaxf(acc, 0.f);
  }
  __syncthreads();
  if (tid == 0) {
    float acc = p2_b[0];
    for (int i = 0; i < H_OUT; ++i) acc += p2_W[i] * h1[i];
    out[b] = acc;
  }
}

// ---------------------------------------------------------------------------
extern "C" void kernel_launch(void* const* d_in, const int* in_sizes, int n_in,
                              void* d_out, int out_size, void* d_ws, size_t ws_size,
                              hipStream_t stream) {
  (void)in_sizes; (void)n_in; (void)out_size; (void)ws_size;
  const int*   rna       = (const int*)d_in[0];
  const int*   tis_id    = (const int*)d_in[1];
  const int*   slen      = (const int*)d_in[2];
  const float* seq_W     = (const float*)d_in[3];
  const float* tis_W     = (const float*)d_in[4];
  const float* in_proj_W = (const float*)d_in[5];
  const float* conv_w    = (const float*)d_in[6];
  const float* conv_b    = (const float*)d_in[7];
  const float* x_proj_W  = (const float*)d_in[8];
  const float* dt_proj_W = (const float*)d_in[9];
  const float* dt_proj_b = (const float*)d_in[10];
  const float* A_log     = (const float*)d_in[11];
  const float* D_skip    = (const float*)d_in[12];
  const float* out_proj_W= (const float*)d_in[13];
  const float* p1_W      = (const float*)d_in[14];
  const float* p1_b      = (const float*)d_in[15];
  const float* p2_W      = (const float*)d_in[16];
  const float* p2_b      = (const float*)d_in[17];
  float* out = (float*)d_out;

  char* ws = (char*)d_ws;
  size_t off = 0;
  auto carve = [&](size_t bytes) -> char* {
    char* p = ws + off;
    off = (off + bytes + 255) & ~(size_t)255;
    return p;
  };
  const size_t M = (size_t)B_ * L_;
  float*          seq_scale = (float*)carve(VOCAB * sizeof(float));
  float*          tis_scale = (float*)carve(30 * sizeof(float));
  unsigned short* Xbf       = (unsigned short*)carve(M * D_MODEL * sizeof(unsigned short));
  unsigned short* Wbf       = (unsigned short*)carve((size_t)D_INNER * D_MODEL * sizeof(unsigned short));
  float*          xm_raw    = (float*)carve(M * D_INNER * sizeof(float));
  float*          xm        = (float*)carve(M * D_INNER * sizeof(float));
  float*          dtBC      = (float*)carve(M * XD * sizeof(float));
  float*          y_last    = (float*)carve((size_t)B_ * D_INNER * sizeof(float));

  norms_kernel<<<1, 128, 0, stream>>>(seq_W, tis_W, seq_scale, tis_scale);
  wcast_kernel<<<D_INNER, 320, 0, stream>>>(in_proj_W, Wbf);
  embed_kernel<<<(int)M, 320, 0, stream>>>(rna, tis_id, seq_W, tis_W, seq_scale, tis_scale, Xbf);
  inproj_wmma_kernel<<<dim3(D_INNER / 128, (int)(M / 16)), 256, 0, stream>>>(Xbf, Wbf, xm_raw);
  conv_silu_kernel<<<(int)M, D_INNER, 0, stream>>>(xm_raw, conv_w, conv_b, xm);
  xproj_kernel<<<(int)M, 64, 0, stream>>>(xm, x_proj_W, dtBC);
  scan_kernel<<<dim3(B_, D_INNER / 128), 128, 0, stream>>>(dtBC, xm, dt_proj_W, dt_proj_b,
                                                           A_log, D_skip, slen, y_last);
  head_kernel<<<B_, D_INNER, 0, stream>>>(rna, tis_id, slen, seq_W, tis_W, seq_scale, tis_scale,
                                          in_proj_W, y_last, out_proj_W,
                                          p1_W, p1_b, p2_W, p2_b, out);
}
